// Attention_7541962571806
// MI455X (gfx1250) — matrix operand order repaired and checked by reading, not compiled, not run on hardware
//
#include <hip/hip_runtime.h>

typedef __attribute__((ext_vector_type(16))) _Float16 v16h;
typedef __attribute__((ext_vector_type(8)))  float    v8f;

union AFrag { v16h v; unsigned u[8]; _Float16 h[16]; };
union H16   { uint4 q[2]; _Float16 h[16]; };

static constexpr float ATT_SCALE = 0.125f;  // 64^-0.5

// --- CDNA5 async global->LDS copy (ASYNCcnt tracked), 16B per lane ---------
__device__ __forceinline__ void async_copy_b128(const void* gaddr, void* lds) {
  asm volatile("global_load_async_to_lds_b128 %0, %1, off"
               :: "v"((unsigned)(size_t)lds), "v"(gaddr)
               : "memory");
}
__device__ __forceinline__ void wait_async0() {
  asm volatile("s_wait_asynccnt 0x0" ::: "memory");
}

// ---------------------------------------------------------------------------
// LayerNorm (fp32 in) -> f16 out.  One wave32 per 1024-element row.
// ---------------------------------------------------------------------------
__global__ __launch_bounds__(256) void ln_kernel(
    const float* __restrict__ x, const float* __restrict__ gamma,
    const float* __restrict__ beta, _Float16* __restrict__ out)
{
  const int row  = blockIdx.x * 8 + (threadIdx.x >> 5);
  const int lane = threadIdx.x & 31;
  const float4* xv = reinterpret_cast<const float4*>(x + (size_t)row * 1024);
  float4 vals[8];
  float s = 0.f, s2 = 0.f;
#pragma unroll
  for (int j = 0; j < 8; ++j) {
    float4 v = xv[lane * 8 + j];
    vals[j] = v;
    s  += v.x + v.y + v.z + v.w;
    s2 += v.x*v.x + v.y*v.y + v.z*v.z + v.w*v.w;
  }
#pragma unroll
  for (int m = 16; m; m >>= 1) {
    s  += __shfl_xor(s,  m, 32);
    s2 += __shfl_xor(s2, m, 32);
  }
  const float mean = s * (1.f / 1024.f);
  const float var  = s2 * (1.f / 1024.f) - mean * mean;
  const float rstd = rsqrtf(var + 1e-5f);
  _Float16* orow = out + (size_t)row * 1024;
#pragma unroll
  for (int j = 0; j < 8; ++j) {
    const int idx = lane * 32 + j * 4;
    float4 v = vals[j];
    orow[idx + 0] = (_Float16)((v.x - mean) * rstd * gamma[idx + 0] + beta[idx + 0]);
    orow[idx + 1] = (_Float16)((v.y - mean) * rstd * gamma[idx + 1] + beta[idx + 1]);
    orow[idx + 2] = (_Float16)((v.z - mean) * rstd * gamma[idx + 2] + beta[idx + 2]);
    orow[idx + 3] = (_Float16)((v.w - mean) * rstd * gamma[idx + 3] + beta[idx + 3]);
  }
}

// ---------------------------------------------------------------------------
// fp32 -> f16 conversion (weights).  n must be a multiple of 4.
// ---------------------------------------------------------------------------
__global__ __launch_bounds__(256) void cvt_f32_f16(
    const float* __restrict__ in, _Float16* __restrict__ out, int n)
{
  const int i = (blockIdx.x * 256 + threadIdx.x) * 4;
  if (i < n) {
    float4 v = *reinterpret_cast<const float4*>(in + i);
    out[i + 0] = (_Float16)v.x;
    out[i + 1] = (_Float16)v.y;
    out[i + 2] = (_Float16)v.z;
    out[i + 3] = (_Float16)v.w;
  }
}

// ---------------------------------------------------------------------------
// f16 WMMA GEMM:  C[M,N] = A[M,K] * B[K,N]
// Block tile 64x128, 8 waves (2 M x 4 N), each wave 32x32 (2x2 WMMA tiles).
// Double-buffered LDS fed by global_load_async_to_lds_b128 (ASYNCcnt).
// ---------------------------------------------------------------------------
#define GA_STRIDE 48   // 64 rows x (32+16) halves
#define GB_STRIDE 144  // 32 rows x (128+16) halves

__global__ __launch_bounds__(256) void gemm_f16(
    const _Float16* __restrict__ A, const _Float16* __restrict__ B,
    _Float16* __restrict__ Cf16, float* __restrict__ Cf32,
    const float* __restrict__ bias, int M, int N, int K, int out_f32)
{
  __shared__ __align__(32) _Float16 Alds[2][64 * GA_STRIDE];
  __shared__ __align__(32) _Float16 Blds[2][32 * GB_STRIDE];

  const int tid  = threadIdx.x;
  const int lane = tid & 31;
  const int wv   = tid >> 5;   // 0..7
  const int wm   = wv >> 2;    // 0..1 (M)
  const int wn   = wv & 3;     // 0..3 (N)
  const int bm   = blockIdx.x * 64;
  const int bn   = blockIdx.y * 128;

  v8f acc[2][2] = {};

  const int arow = tid >> 2;            // 0..63
  const int akp  = (tid & 3) * 8;       // 0,8,16,24
  const int brow = tid >> 3;            // 0..31
  const int bnp  = (tid & 7) * 16;      // 0..112

  // prologue: async-stage tile 0 into buffer 0
  async_copy_b128(A + (size_t)(bm + arow) * K + akp,
                  &Alds[0][arow * GA_STRIDE + akp]);
  async_copy_b128(B + (size_t)brow * N + bn + bnp,
                  &Blds[0][brow * GB_STRIDE + bnp]);
  async_copy_b128(B + (size_t)brow * N + bn + bnp + 8,
                  &Blds[0][brow * GB_STRIDE + bnp + 8]);
  wait_async0();
  __syncthreads();

  int buf = 0;
  for (int k0 = 0; k0 < K; k0 += 32) {
    // async-stage next K tile into the other buffer while we compute
    if (k0 + 32 < K) {
      const int nb = buf ^ 1;
      async_copy_b128(A + (size_t)(bm + arow) * K + k0 + 32 + akp,
                      &Alds[nb][arow * GA_STRIDE + akp]);
      async_copy_b128(B + (size_t)(k0 + 32 + brow) * N + bn + bnp,
                      &Blds[nb][brow * GB_STRIDE + bnp]);
      async_copy_b128(B + (size_t)(k0 + 32 + brow) * N + bn + bnp + 8,
                      &Blds[nb][brow * GB_STRIDE + bnp + 8]);
    }

    // --- A fragments (16x32 f16 layout: M=lane%16, K split by lane-half) ---
    AFrag a[2];
#pragma unroll
    for (int mt = 0; mt < 2; ++mt) {
      const int r  = wm * 32 + mt * 16 + (lane & 15);
      const int hb = (lane >> 4) * 8;
#pragma unroll
      for (int v = 0; v < 8; ++v) {
        const int kk = (v >> 2) * 16 + hb + (v & 3) * 2;
        a[mt].u[v] =
            *reinterpret_cast<const unsigned*>(&Alds[buf][r * GA_STRIDE + kk]);
      }
    }
    // --- B fragments (32x16 f16 layout: N=lane%16, K split by lane-half) ---
#pragma unroll
    for (int nt = 0; nt < 2; ++nt) {
      v16h bf;
#pragma unroll
      for (int s = 0; s < 16; ++s)
        bf[s] = Blds[buf][((s >> 3) * 16 + (lane >> 4) * 8 + (s & 7)) * GB_STRIDE + wn * 32 + nt * 16 + (lane & 15)];
#pragma unroll
      for (int mt = 0; mt < 2; ++mt) {
        acc[mt][nt] = __builtin_amdgcn_wmma_f32_16x16x32_f16(
            false, a[mt].v, false, bf, (short)0, acc[mt][nt], false, false);
      }
    }

    wait_async0();      // my async writes to buf^1 have landed
    __syncthreads();    // everyone's have; everyone's reads of buf retired
    buf ^= 1;
  }

  // --- store (C layout: N=lane%16, M=r+8*(lane>=16)) ---
  const int rb = (lane >> 4) * 8;
  const int nc = lane & 15;
#pragma unroll
  for (int mt = 0; mt < 2; ++mt) {
#pragma unroll
    for (int nt = 0; nt < 2; ++nt) {
      const int col = bn + wn * 32 + nt * 16 + nc;
#pragma unroll
      for (int r = 0; r < 8; ++r) {
        const size_t row = (size_t)(bm + wm * 32 + mt * 16 + rb + r);
        const float val = acc[mt][nt][r];
        if (out_f32) Cf32[row * N + col] = val + bias[col];
        else         Cf16[row * N + col] = (_Float16)val;
      }
    }
  }
}

// ---------------------------------------------------------------------------
// Flash attention over qkv (f16, rows = [Q|K|V] of 1024 each).
// grid.x = b*heads (64), grid.y = query blocks of 128.  8 waves x 16 q-rows.
// K staged TRANSPOSED in LDS (Kt[d][key]) ; Q*K^T B-fragments are gathered as
// 16 strided 16-bit LDS reads.  V staged via async global->LDS.  Online softmax.
// ---------------------------------------------------------------------------
#define KT_STRIDE 80   // 64 keys + 16 pad halves (rows stay 32B aligned)
#define KV_STRIDE 80   // 64 d + 16 pad halves
#define P_STRIDE  72   // 64 + 8 halves

__global__ __launch_bounds__(256) void attn_kernel(
    const _Float16* __restrict__ qkv, _Float16* __restrict__ out)
{
  __shared__ __align__(32) _Float16 Kt[64 * KT_STRIDE];  // [d][key]
  __shared__ __align__(32) _Float16 Vl[64 * KV_STRIDE];  // [key][d]
  __shared__ __align__(16) _Float16 Pl[8][16 * P_STRIDE];

  const int bh    = blockIdx.x;
  const int batch = bh >> 4;
  const int head  = bh & 15;
  const int tid   = threadIdx.x;
  const int lane  = tid & 31;
  const int wv    = tid >> 5;

  const size_t rs   = 3072;
  const size_t base = (size_t)batch * 2048 * rs;
  const int qoff = head * 64;
  const int koff = 1024 + head * 64;
  const int voff = 2048 + head * 64;

  // --- load Q fragments (d=64 -> two K32 chunks), fold in softmax scale ---
  const int qrow0 = blockIdx.y * 128 + wv * 16;
  AFrag qa[2];
  {
    const int r  = qrow0 + (lane & 15);
    const int hb = (lane >> 4) * 8;
    const _Float16* qr = qkv + base + (size_t)r * rs + qoff;
#pragma unroll
    for (int c = 0; c < 2; ++c) {
#pragma unroll
      for (int v = 0; v < 8; ++v) {
        const int kk = c * 32 + (v >> 2) * 16 + hb + (v & 3) * 2;
        qa[c].h[2 * v + 0] = (_Float16)((float)qr[kk + 0] * ATT_SCALE);
        qa[c].h[2 * v + 1] = (_Float16)((float)qr[kk + 1] * ATT_SCALE);
      }
    }
  }

  float mrow[8], lsum[8];
  v8f o[4] = {};
#pragma unroll
  for (int r = 0; r < 8; ++r) { mrow[r] = -1e30f; lsum[r] = 0.f; }

  const int ki = tid >> 2;            // key row 0..63 for staging
  const int kd = (tid & 3) * 16;      // 16 halves per thread

  for (int kb = 0; kb < 2048; kb += 64) {
    __syncthreads();   // previous tile fully consumed before overwrite
    {
      const _Float16* kr = qkv + base + (size_t)(kb + ki) * rs;
      // V: straight async copy into [key][d]
      async_copy_b128(kr + voff + kd,     &Vl[ki * KV_STRIDE + kd]);
      async_copy_b128(kr + voff + kd + 8, &Vl[ki * KV_STRIDE + kd + 8]);
      // K: load 16 halves along d, scatter transposed into Kt[d][key]
      H16 kk;
      kk.q[0] = *reinterpret_cast<const uint4*>(kr + koff + kd);
      kk.q[1] = *reinterpret_cast<const uint4*>(kr + koff + kd + 8);
#pragma unroll
      for (int j = 0; j < 16; ++j)
        Kt[(kd + j) * KT_STRIDE + ki] = kk.h[j];
      // prefetch next key block
      if (kb + 64 < 2048) {
        __builtin_prefetch(kr + 64 * rs + koff + kd, 0, 1);
        __builtin_prefetch(kr + 64 * rs + voff + kd, 0, 1);
      }
    }
    wait_async0();
    __syncthreads();

    // --- S = Q * K^T for 4 key sub-tiles of 16 ---
    v8f st[4];
#pragma unroll
    for (int kt = 0; kt < 4; ++kt) {
      v16h bf0, bf1;
#pragma unroll
      for (int s = 0; s < 16; ++s) {
        const int d = (s >> 3) * 16 + (lane >> 4) * 8 + (s & 7);
        bf0[s] = Kt[d * KT_STRIDE + kt * 16 + (lane & 15)];
        bf1[s] = Kt[(32 + d) * KT_STRIDE + kt * 16 + (lane & 15)];
      }
      v8f z = {};
      z = __builtin_amdgcn_wmma_f32_16x16x32_f16(false, qa[0].v, false, bf0,
                                                 (short)0, z, false, false);
      z = __builtin_amdgcn_wmma_f32_16x16x32_f16(false, qa[1].v, false, bf1,
                                                 (short)0, z, false, false);
      st[kt] = z;
    }

    // --- online softmax (half-wave reductions: rows split by lane group) ---
    float mnew[8], alpha[8], rsum[8];
#pragma unroll
    for (int r = 0; r < 8; ++r) {
      float t = fmaxf(fmaxf(st[0][r], st[1][r]), fmaxf(st[2][r], st[3][r]));
#pragma unroll
      for (int msk = 1; msk <= 8; msk <<= 1) t = fmaxf(t, __shfl_xor(t, msk, 32));
      mnew[r]  = fmaxf(mrow[r], t);
      alpha[r] = __expf(mrow[r] - mnew[r]);
      rsum[r]  = 0.f;
    }
#pragma unroll
    for (int kt = 0; kt < 4; ++kt) {
#pragma unroll
      for (int r = 0; r < 8; ++r) {
        const float p = __expf(st[kt][r] - mnew[r]);
        st[kt][r] = p;
        rsum[r] += p;
      }
    }
#pragma unroll
    for (int r = 0; r < 8; ++r) {
#pragma unroll
      for (int msk = 1; msk <= 8; msk <<= 1) rsum[r] += __shfl_xor(rsum[r], msk, 32);
      lsum[r] = lsum[r] * alpha[r] + rsum[r];
      mrow[r] = mnew[r];
    }
#pragma unroll
    for (int dt = 0; dt < 4; ++dt)
#pragma unroll
      for (int r = 0; r < 8; ++r) o[dt][r] *= alpha[r];

    // --- P: C-layout -> f16 in per-wave LDS -> A fragments; O += P*V ---
    {
      _Float16* P = &Pl[wv][0];
      const int rb = (lane >> 4) * 8;
      const int nc = lane & 15;
#pragma unroll
      for (int kt = 0; kt < 4; ++kt)
#pragma unroll
        for (int r = 0; r < 8; ++r)
          P[(rb + r) * P_STRIDE + kt * 16 + nc] = (_Float16)st[kt][r];

      const int rr = lane & 15;
      const int hb = (lane >> 4) * 8;
#pragma unroll
      for (int c = 0; c < 2; ++c) {      // key chunks of 32
        AFrag pa;
#pragma unroll
        for (int v = 0; v < 8; ++v) {
          const int kk = c * 32 + (v >> 2) * 16 + hb + (v & 3) * 2;
          pa.u[v] = *reinterpret_cast<const unsigned*>(&P[rr * P_STRIDE + kk]);
        }
#pragma unroll
        for (int dt = 0; dt < 4; ++dt) {
          v16h vf;
#pragma unroll
          for (int s = 0; s < 16; ++s)
            vf[s] = Vl[(c * 32 + (s >> 3) * 16 + (lane >> 4) * 8 + (s & 7)) * KV_STRIDE + dt * 16 + (lane & 15)];
          o[dt] = __builtin_amdgcn_wmma_f32_16x16x32_f16(
              false, pa.v, false, vf, (short)0, o[dt], false, false);
        }
      }
    }
  }

  // --- epilogue: normalize, write attn_out f16 [8192,1024] ---
  const int rb = (lane >> 4) * 8;
  const int nc = lane & 15;
  float inv_l[8];
#pragma unroll
  for (int r = 0; r < 8; ++r) inv_l[r] = 1.f / lsum[r];
#pragma unroll
  for (int dt = 0; dt < 4; ++dt) {
#pragma unroll
    for (int r = 0; r < 8; ++r) {
      const size_t grow = (size_t)batch * 2048 + qrow0 + rb + r;
      out[grow * 1024 + head * 64 + dt * 16 + nc] =
          (_Float16)(o[dt][r] * inv_l[r]);
    }
  }
}

// ---------------------------------------------------------------------------
extern "C" void kernel_launch(void* const* d_in, const int* in_sizes, int n_in,
                              void* d_out, int out_size, void* d_ws, size_t ws_size,
                              hipStream_t stream) {
  const float* x     = (const float*)d_in[0];
  const float* gamma = (const float*)d_in[1];
  const float* beta  = (const float*)d_in[2];
  const float* wqkv  = (const float*)d_in[3];
  const float* wout  = (const float*)d_in[4];
  const float* bout  = (const float*)d_in[5];
  float* out = (float*)d_out;

  const size_t ROWS = 4 * 2048;        // 8192
  _Float16* xn   = (_Float16*)d_ws;                  // 8192 x 1024
  _Float16* wq16 = xn   + ROWS * 1024;               // 1024 x 3072
  _Float16* wo16 = wq16 + (size_t)1024 * 3072;       // 1024 x 1024
  _Float16* qkv  = wo16 + (size_t)1024 * 1024;       // 8192 x 3072
  _Float16* ao   = qkv  + ROWS * 3072;               // 8192 x 1024

  // 1) LayerNorm -> f16
  ln_kernel<<<dim3(ROWS / 8), dim3(256), 0, stream>>>(x, gamma, beta, xn);

  // 2) weights -> f16
  cvt_f32_f16<<<dim3((1024 * 3072) / (256 * 4)), dim3(256), 0, stream>>>(
      wqkv, wq16, 1024 * 3072);
  cvt_f32_f16<<<dim3((1024 * 1024) / (256 * 4)), dim3(256), 0, stream>>>(
      wout, wo16, 1024 * 1024);

  // 3) QKV projection: [8192,1024] x [1024,3072] -> f16
  gemm_f16<<<dim3(ROWS / 64, 3072 / 128), dim3(256), 0, stream>>>(
      xn, wq16, qkv, nullptr, nullptr, (int)ROWS, 3072, 1024, 0);

  // 4) Flash attention per (batch, head)
  attn_kernel<<<dim3(64, 2048 / 128), dim3(256), 0, stream>>>(qkv, ao);

  // 5) Output projection + bias: [8192,1024] x [1024,1024] -> f32 d_out
  gemm_f16<<<dim3(ROWS / 64, 1024 / 128), dim3(256), 0, stream>>>(
      ao, wo16, nullptr, out, bout, (int)ROWS, 1024, 1024, 1);
}
